// CRAU_16544214024674
// MI455X (gfx1250) — matrix operand order repaired and checked by reading, not compiled, not run on hardware
//
#include <hip/hip_runtime.h>

#define C_CH 256
#define HS   128
#define WSZ  128
#define PS   (HS * WSZ)   // 16384 pixels in src image
#define HF   64
#define WF   64
#define PF   (HF * WF)    // 4096 pixels in feat image
#define NB   4

typedef __attribute__((ext_vector_type(16))) _Float16 v16h;
typedef __attribute__((ext_vector_type(2)))  __fp16   v2fp;   // cvt_pkrtz return type
typedef __attribute__((ext_vector_type(8)))  float    v8f;

union HFrag { v16h v; v2fp p[8]; };

// For 128-space index t (qc row/col or output row/col), enumerate the valid
// (kernel-offset i, 64-space index o) pairs satisfying t + 1 == i + 2*o.
__device__ __forceinline__ int pairs_for(int t, int* ii, int* oo) {
  int n = 0;
  if (t & 1) {
    if (t < 127) { ii[n] = 0; oo[n] = (t + 1) >> 1; ++n; }
    ii[n] = 2; oo[n] = (t - 1) >> 1; ++n;
  } else {
    ii[n] = 1; oo[n] = t >> 1; ++n;
  }
  return n;
}

// A fragment (16x32 f16): lane m=lane&15, hi=lane>>4.
// element e -> K = e + 16*hi (matches B fragment).
// Consecutive element pairs are consecutive K -> pack with v_cvt_pk_rtz_f16_f32.
__device__ __forceinline__ v16h load_a_frag(const float* __restrict__ wrow, int hi) {
  HFrag a;
  float4 f0 = *reinterpret_cast<const float4*>(wrow + 16 * hi);
  float4 f1 = *reinterpret_cast<const float4*>(wrow + 16 * hi + 4);
  float4 f2 = *reinterpret_cast<const float4*>(wrow + 16 * hi + 8);
  float4 f3 = *reinterpret_cast<const float4*>(wrow + 16 * hi + 12);
  a.p[0] = __builtin_amdgcn_cvt_pkrtz(f0.x, f0.y);
  a.p[1] = __builtin_amdgcn_cvt_pkrtz(f0.z, f0.w);
  a.p[2] = __builtin_amdgcn_cvt_pkrtz(f1.x, f1.y);
  a.p[3] = __builtin_amdgcn_cvt_pkrtz(f1.z, f1.w);
  a.p[4] = __builtin_amdgcn_cvt_pkrtz(f2.x, f2.y);
  a.p[5] = __builtin_amdgcn_cvt_pkrtz(f2.z, f2.w);
  a.p[6] = __builtin_amdgcn_cvt_pkrtz(f3.x, f3.y);
  a.p[7] = __builtin_amdgcn_cvt_pkrtz(f3.z, f3.w);
  return a.v;
}

// B fragment (32x16 f16): element e -> K = e + 16*hi, column n = lane&15.
// xp points at X[(k0 + 16*hi)*PIX + col]; stride PIX between K rows.
template <int PIX>
__device__ __forceinline__ v16h load_b_frag(const float* __restrict__ xp) {
  HFrag b;
#pragma unroll
  for (int j = 0; j < 8; ++j)
    b.p[j] = __builtin_amdgcn_cvt_pkrtz(xp[(size_t)(2 * j) * PIX],
                                        xp[(size_t)(2 * j + 1) * PIX]);
  return b.v;
}

// 2x2 register-tiled GEMM core: D[32ch x 32px] += W[32ch x 256] * X[256 x 32px]
template <int PIX>
__device__ __forceinline__ void gemm_tile(const float* __restrict__ X,
                                          const float* __restrict__ W,
                                          int o0, int p0, int m, int hi,
                                          v8f acc[2][2]) {
  for (int k0 = 0; k0 < C_CH; k0 += 32) {
    v16h a0 = load_a_frag(W + (o0 + m) * C_CH + k0, hi);
    v16h a1 = load_a_frag(W + (o0 + 16 + m) * C_CH + k0, hi);
    const float* xb = X + (size_t)(k0 + 16 * hi) * PIX + p0 + m;
    v16h b0 = load_b_frag<PIX>(xb);
    v16h b1 = load_b_frag<PIX>(xb + 16);
    acc[0][0] = __builtin_amdgcn_wmma_f32_16x16x32_f16(false, a0, false, b0, (short)0,
                                                       acc[0][0], false, false);
    acc[0][1] = __builtin_amdgcn_wmma_f32_16x16x32_f16(false, a0, false, b1, (short)0,
                                                       acc[0][1], false, false);
    acc[1][0] = __builtin_amdgcn_wmma_f32_16x16x32_f16(false, a1, false, b0, (short)0,
                                                       acc[1][0], false, false);
    acc[1][1] = __builtin_amdgcn_wmma_f32_16x16x32_f16(false, a1, false, b1, (short)0,
                                                       acc[1][1], false, false);
  }
}

// ---------------- vc = Wv @ feat + bv  -> workspace ----------------
// Block: 256 thr = 8 waves; block tile 64ch x 128px; wave tile 32ch x 32px.
__global__ __launch_bounds__(256) void vc_gemm(const float* __restrict__ feat,
                                               const float* __restrict__ Wv,
                                               const float* __restrict__ bv,
                                               float* __restrict__ vc) {
  int blk = blockIdx.x;
  int pgrp = blk % (PF / 128); blk /= (PF / 128);
  int ot = blk & 3;            blk >>= 2;
  int b = blk;
  int wave = threadIdx.x >> 5;
  int lane = threadIdx.x & 31;
  int m = lane & 15;
  int hi = lane >> 4;
  int o0 = ot * 64 + (wave >> 2) * 32;
  int p0 = pgrp * 128 + (wave & 3) * 32;
  const float* X = feat + (size_t)b * C_CH * PF;

  v8f acc[2][2] = {};
  gemm_tile<PF>(X, Wv, o0, p0, m, hi, acc);

#pragma unroll
  for (int ai = 0; ai < 2; ++ai)
#pragma unroll
    for (int r = 0; r < 8; ++r) {
      int c = o0 + ai * 16 + r + 8 * hi;
      float bias = bv[c];
      float* orow = vc + ((size_t)b * C_CH + c) * PF + p0 + m;
      orow[0]  = acc[ai][0][r] + bias;
      orow[16] = acc[ai][1][r] + bias;
    }
}

// ------- fused: qc = Wq @ src + bq, scatter-reduce into S[b,c,9] -------
__global__ __launch_bounds__(256) void qS_gemm(const float* __restrict__ src,
                                               const float* __restrict__ Wq,
                                               const float* __restrict__ bq,
                                               const float* __restrict__ feat,
                                               float* __restrict__ S) {
  __shared__ float sS[64 * 9];
  for (int t = threadIdx.x; t < 576; t += blockDim.x) sS[t] = 0.f;
  __syncthreads();

  int blk = blockIdx.x;
  int pgrp = blk % (PS / 128); blk /= (PS / 128);
  int ot = blk & 3;            blk >>= 2;
  int b = blk;
  int wave = threadIdx.x >> 5;
  int lane = threadIdx.x & 31;
  int m = lane & 15;
  int hi = lane >> 4;
  int chsub = wave >> 2;
  int o0 = ot * 64 + chsub * 32;
  int p0 = pgrp * 128 + (wave & 3) * 32;
  const float* X = src + (size_t)b * C_CH * PS;

  v8f acc[2][2] = {};
  gemm_tile<PS>(X, Wq, o0, p0, m, hi, acc);

  // Both pixel columns (p0+m, p0+m+16) sit in the same image row.
  int p = p0 + m;
  int h = p >> 7;
  int iy[2], oyv[2];
  int ny = pairs_for(h, iy, oyv);
  int jx[2][2], oxv[2][2], nx[2];
  nx[0] = pairs_for(p & 127, jx[0], oxv[0]);
  nx[1] = pairs_for((p & 127) + 16, jx[1], oxv[1]);

#pragma unroll
  for (int ai = 0; ai < 2; ++ai)
#pragma unroll
    for (int r = 0; r < 8; ++r) {
      int cl = chsub * 32 + ai * 16 + r + 8 * hi;  // channel within block [0,64)
      int c  = ot * 64 + cl;
      const float* F = feat + ((size_t)b * C_CH + c) * PF;
      float bias = bq[c];
#pragma unroll
      for (int bi = 0; bi < 2; ++bi) {
        float qv = acc[ai][bi][r] + bias;  // qc value (padding handled by skipping)
        for (int u = 0; u < ny; ++u)
          for (int t2 = 0; t2 < nx[bi]; ++t2)
            atomicAdd(&sS[cl * 9 + iy[u] * 3 + jx[bi][t2]],
                      qv * F[oyv[u] * WF + oxv[bi][t2]]);
      }
    }
  __syncthreads();
  for (int t = threadIdx.x; t < 576; t += blockDim.x) {
    int cl = t / 9, k = t % 9;
    atomicAdd(&S[((size_t)b * C_CH + ot * 64 + cl) * 9 + k], sS[t]);
  }
}

// ---------------- softmax over the 9 logits (in place) ----------------
__global__ void softmax9(float* __restrict__ S) {
  int idx = blockIdx.x * blockDim.x + threadIdx.x;
  if (idx >= NB * C_CH) return;
  const float scale = 1.0f / 64.0f;  // 1/sqrt(4096)
  float v[9], mx = -1e30f;
#pragma unroll
  for (int k = 0; k < 9; ++k) { v[k] = S[idx * 9 + k] * scale; mx = fmaxf(mx, v[k]); }
  float sum = 0.f;
#pragma unroll
  for (int k = 0; k < 9; ++k) { v[k] = __expf(v[k] - mx); sum += v[k]; }
  float inv = 1.0f / sum;
#pragma unroll
  for (int k = 0; k < 9; ++k) S[idx * 9 + k] = v[k] * inv;
}

// --------- out = fold(A ⊗ vc) * src  (≤4 terms per output pixel) ---------
__global__ __launch_bounds__(256) void fold_mul(const float* __restrict__ src,
                                                const float* __restrict__ vc,
                                                const float* __restrict__ A,
                                                float* __restrict__ out) {
  size_t idx = (size_t)blockIdx.x * blockDim.x + threadIdx.x;
  int p = (int)(idx & (PS - 1));
  int bc = (int)(idx >> 14);   // PS = 2^14
  int y = p >> 7, x = p & 127;
  int iy[2], oy[2], jx[2], ox[2];
  int ny = pairs_for(y, iy, oy);
  int nx = pairs_for(x, jx, ox);
  const float* Vp = vc + (size_t)bc * PF;
  const float* Ap = A + (size_t)bc * 9;
  float s = 0.f;
  for (int ai = 0; ai < ny; ++ai)
    for (int bi = 0; bi < nx; ++bi)
      s += Ap[iy[ai] * 3 + jx[bi]] * Vp[oy[ai] * WF + ox[bi]];
  out[idx] = s * src[idx];
}

__global__ void zero_buf(float* __restrict__ p, int n) {
  int i = blockIdx.x * blockDim.x + threadIdx.x;
  if (i < n) p[i] = 0.f;
}

extern "C" void kernel_launch(void* const* d_in, const int* in_sizes, int n_in,
                              void* d_out, int out_size, void* d_ws, size_t ws_size,
                              hipStream_t stream) {
  (void)in_sizes; (void)n_in; (void)out_size; (void)ws_size;
  const float* feat = (const float*)d_in[0];  // [4,256,64,64]
  const float* src  = (const float*)d_in[1];  // [4,256,128,128]
  const float* Wq   = (const float*)d_in[2];  // [256,256]
  const float* bq   = (const float*)d_in[3];  // [256]
  const float* Wv   = (const float*)d_in[4];  // [256,256]
  const float* bv   = (const float*)d_in[5];  // [256]
  float* out = (float*)d_out;                 // [4,256,128,128]

  float* vc = (float*)d_ws;                           // NB*C*PF floats (16.8 MB)
  float* S  = vc + (size_t)NB * C_CH * PF;            // NB*C*9 floats (36 KB)

  int nS = NB * C_CH * 9;
  zero_buf<<<(nS + 255) / 256, 256, 0, stream>>>(S, nS);
  vc_gemm<<<NB * 4 * (PF / 128), 256, 0, stream>>>(feat, Wv, bv, vc);
  qS_gemm<<<NB * 4 * (PS / 128), 256, 0, stream>>>(src, Wq, bq, feat, S);
  softmax9<<<(NB * C_CH + 255) / 256, 256, 0, stream>>>(S);
  fold_mul<<<(NB * C_CH * PS) / 256, 256, 0, stream>>>(src, vc, S, out);
}
